// MultiHeadSelfAttention_14757507629346
// MI455X (gfx1250) — matrix operand order repaired and checked by reading, not compiled, not run on hardware
//
#include <hip/hip_runtime.h>
#include <hip/hip_bf16.h>
#include <stdint.h>

typedef __bf16 bf16;
typedef __bf16 v16bf __attribute__((ext_vector_type(16)));
typedef __bf16 v8bf  __attribute__((ext_vector_type(8)));
typedef __bf16 v4bf  __attribute__((ext_vector_type(4)));
typedef float  v8f   __attribute__((ext_vector_type(8)));
typedef unsigned int uint4v __attribute__((ext_vector_type(4)));
typedef int int8v __attribute__((ext_vector_type(8)));

#define D_MODEL 1024
#define N_HEADS 16
#define HEAD    64
#define BATCH   4
#define SEQ     2048
#define ROWS    (BATCH * SEQ)   // 8192

union FragU { v16bf v; v8bf h[2]; };

// 16-bit A fragment (16x32, row-major source, stride ld elements).
// Per CDNA5 ISA 7.12.2: lanes 0-15 hold row m, K = {0..7, 16..23};
// lanes 16-31 hold row m, K = {8..15, 24..31}.  -> two 16B loads per lane.
__device__ __forceinline__ v16bf load_a_frag(const bf16* base, int ld, int lane) {
  const int m = lane & 15, hf = lane >> 4;
  const bf16* row = base + (size_t)m * ld + 8 * hf;
  FragU f;
  f.h[0] = *(const v8bf*)(row);
  f.h[1] = *(const v8bf*)(row + 16);
  return f.v;
}

// 16-bit B fragment (32x16). Lane n = column n (lane&15); lanes 0-15 hold
// K={0..7, 16..23}, lanes 16-31 hold K={8..15, 24..31}. Source is "column-contiguous": data of
// column n at base + n*ld, contiguous along K.  -> two 16B loads per lane.
__device__ __forceinline__ v16bf load_b_frag(const bf16* base, int ld, int lane) {
  const int nn = lane & 15, hf = lane >> 4;
  const bf16* p = base + (size_t)nn * ld + 8 * hf;
  FragU f;
  f.h[0] = *(const v8bf*)(p);
  f.h[1] = *(const v8bf*)(p + 16);
  return f.v;
}

__device__ __forceinline__ v8f wmma_bf16(v16bf a, v16bf b, v8f c) {
  return __builtin_amdgcn_wmma_f32_16x16x32_bf16(false, a, false, b, (short)0, c,
                                                 false, false);
}

// Async memory->LDS copy, 16B per active lane (GLOBAL_LOAD_ASYNC_TO_LDS_B128,
// tracked by ASYNCcnt). lds_addr = per-lane LDS byte address (VDST), gaddr =
// per-lane 64-bit global address (VADDR), saddr = off.
__device__ __forceinline__ void async_ld_lds_b128(uint32_t lds_addr, uint64_t gaddr) {
  asm volatile("global_load_async_to_lds_b128 %0, %1, off"
               :: "v"(lds_addr), "v"(gaddr) : "memory");
}
__device__ __forceinline__ void wait_asynccnt0() {
  asm volatile("s_wait_asynccnt 0x0" ::: "memory");
}

// TDM: 2-D tile load via tensor_load_to_lds (TENSORcnt).  D# per ISA 08 §8.3/8.4.
// data_size = 2 bytes; dims/strides in elements.
__device__ __forceinline__ void tdm_load_tile_2d(uint32_t lds_addr, const void* gp,
                                                 uint32_t tensor_d0, uint32_t tensor_d1,
                                                 uint32_t tile_d0, uint32_t tile_d1,
                                                 uint32_t stride0) {
  const uint64_t ga = (uint64_t)(uintptr_t)gp;
  uint4v g0;
  g0[0] = 1u;                                            // count=1, user D#
  g0[1] = lds_addr;                                      // lds_addr [63:32]
  g0[2] = (uint32_t)ga;                                  // global_addr lo
  g0[3] = ((uint32_t)(ga >> 32) & 0x1FFFFFFu) | (2u << 30);  // addr hi | type=2
  int8v g1;
  g1[0] = (int)(1u << 16);                               // wg_mask=0, data_size=1 (2B)
  g1[1] = (int)((tensor_d0 & 0xFFFFu) << 16);            // tensor_dim0[15:0] @ bit48
  g1[2] = (int)((tensor_d0 >> 16) | ((tensor_d1 & 0xFFFFu) << 16));
  g1[3] = (int)((tensor_d1 >> 16) | (tile_d0 << 16));    // tile_dim0 @ bit112
  g1[4] = (int)(tile_d1 & 0xFFFFu);                      // tile_dim1 (tile_dim2=0)
  g1[5] = (int)stride0;                                  // tensor_dim0_stride lo32
  g1[6] = 0;
  g1[7] = 0;
  asm volatile("tensor_load_to_lds %0, %1" :: "s"(g0), "s"(g1) : "memory");
}

// ---------------------------------------------------------------- narrowing
__global__ void f32_to_bf16_vec(const float* __restrict__ s,
                                bf16* __restrict__ d, int n4) {
  int i = blockIdx.x * blockDim.x + threadIdx.x;
  int stride = gridDim.x * blockDim.x;
  for (; i < n4; i += stride) {
    float4 x = ((const float4*)s)[i];
    v4bf y;
    y[0] = (bf16)x.x; y[1] = (bf16)x.y; y[2] = (bf16)x.z; y[3] = (bf16)x.w;
    ((v4bf*)d)[i] = y;
  }
}

// ------------------------------------------------- fused QKV GEMM + RoPE
// grid.x = 512 row-tiles * 3 outputs * 2 column-halves = 3072, 256 threads.
// Each wave computes a 16x64 strip (one full head) of Q, K or V.
// X k-slice (16x32) is staged into LDS with async global->LDS copies.
__global__ __launch_bounds__(256)
void qkv_rope_kernel(const bf16* __restrict__ X, const bf16* __restrict__ Wq,
                     const bf16* __restrict__ Wk, const bf16* __restrict__ Wv,
                     const long long* __restrict__ tp,
                     bf16* __restrict__ Qo, bf16* __restrict__ Ko,
                     bf16* __restrict__ Vt) {
  __shared__ __align__(16) bf16 xs[16 * 32];
  const int bid = blockIdx.x;
  const int r   = bid & 511;
  const int sel = (bid >> 9) % 3;
  const int hc  = (bid >> 9) / 3;
  const int wave = threadIdx.x >> 5, lane = threadIdx.x & 31;
  const int colbase = hc * 512 + wave * 64;
  const bf16* W = (sel == 0) ? Wq : ((sel == 1) ? Wk : Wv);
  // Generic LDS address: low 32 bits are the LDS byte offset (aperture rule).
  const uint32_t xs_lds = (uint32_t)(uintptr_t)xs;

  v8f acc[4];
#pragma unroll
  for (int g = 0; g < 4; ++g) acc[g] = (v8f){0.f,0.f,0.f,0.f,0.f,0.f,0.f,0.f};

  for (int k0 = 0; k0 < D_MODEL; k0 += 32) {
    __syncthreads();                        // previous tile fully consumed
    // Each wave async-copies its 2 rows (lanes 0-7, 16B each) memory->LDS.
    if (lane < 8) {
      const int rrow = wave * 2 + (lane >> 2);
      const int cseg = (lane & 3) * 8;      // element offset within row
      const uint64_t ga = (uint64_t)(uintptr_t)(
          X + (size_t)(r * 16 + rrow) * D_MODEL + k0 + cseg);
      async_ld_lds_b128(xs_lds + (uint32_t)(rrow * 64 + cseg * 2), ga);
    }
    wait_asynccnt0();
    __syncthreads();

    v16bf a = load_a_frag(xs, 32, lane);
#pragma unroll
    for (int g = 0; g < 4; ++g) {
      const bf16* wb = W + (size_t)(colbase + g * 16) * D_MODEL + k0;
      v16bf b = load_b_frag(wb, D_MODEL, lane);
      acc[g] = wmma_bf16(a, b, acc[g]);
    }
    if (k0 + 32 < D_MODEL)
      __builtin_prefetch(W + (size_t)colbase * D_MODEL + k0 + 32, 0, 1);
  }

  // Epilogue: C layout -> element (m = i + 8*hf, n = lane&15).
  const int n = lane & 15, hf = lane >> 4;
  bf16* dstQK = (sel == 0) ? Qo : Ko;
#pragma unroll
  for (int g = 0; g < 4; ++g) {
    const int e = colbase + g * 16 + n;
    const int h = e >> 6, d = e & 63;
    if (sel < 2) {
      // RoPE: inv_freq = theta^(-(d/2)/32); partner element = adjacent lane.
      const float invf = __expf(-(float)(d >> 1) * 0.2878231366f); // ln(1e4)/32
#pragma unroll
      for (int i = 0; i < 8; ++i) {
        const int row = r * 16 + i + 8 * hf;
        const int b   = row >> 11, s = row & (SEQ - 1);
        const float pos = (float)tp[s];
        float sn, cs;
        __sincosf(pos * invf, &sn, &cs);
        float v = acc[g][i];
        float p = __shfl_xor(v, 1);
        float ov = (d & 1) ? (v * cs + p * sn) : (v * cs - p * sn);
        dstQK[(((size_t)(b * N_HEADS + h) * SEQ + s) * HEAD) + d] = (bf16)ov;
      }
    } else {
      // V stored transposed [B,H,64,S]: 8 consecutive s per lane -> 16B store.
      const int row0 = r * 16 + 8 * hf;
      const int b = row0 >> 11, s0 = row0 & (SEQ - 1);
      v8bf pk;
#pragma unroll
      for (int i = 0; i < 8; ++i) pk[i] = (bf16)acc[g][i];
      *(v8bf*)(Vt + ((size_t)(b * N_HEADS + h) * HEAD + d) * SEQ + s0) = pk;
    }
  }
}

// ------------------------------------------------------- flash attention
// 4 waves/block, one wave per (b,h,16-row q tile).  grid.x = 8192/4 = 2048.
__global__ __launch_bounds__(128)
void flash_attn_kernel(const bf16* __restrict__ Q, const bf16* __restrict__ K,
                       const bf16* __restrict__ Vt, bf16* __restrict__ Oo) {
  __shared__ __align__(16) bf16 pbuf[4][16 * 32];
  const int wave = threadIdx.x >> 5, lane = threadIdx.x & 31;
  const int task = blockIdx.x * 4 + wave;
  const int bh = task >> 7;          // b*16 + h
  const int qt = task & 127;
  const int qbase = qt * 16;
  const int n = lane & 15, hf = lane >> 4;

  const bf16* qb = Q + ((size_t)bh * SEQ + qbase) * HEAD;
  const v16bf aq0 = load_a_frag(qb, HEAD, lane);        // dims 0..31
  const v16bf aq1 = load_a_frag(qb + 32, HEAD, lane);   // dims 32..63

  v8f o[4];
#pragma unroll
  for (int g = 0; g < 4; ++g) o[g] = (v8f){0.f,0.f,0.f,0.f,0.f,0.f,0.f,0.f};
  float mstat[8], lstat[8];
#pragma unroll
  for (int i = 0; i < 8; ++i) { mstat[i] = -3.0e38f; lstat[i] = 0.f; }

  const int kend = qbase + 16;                    // causal: keys < kend
  for (int kb = 0; kb < kend; kb += 32) {
    const bf16* kp = K + ((size_t)bh * SEQ + kb) * HEAD;
    const bf16* vb = Vt + (size_t)bh * HEAD * SEQ + kb;  // V^T rows = dims
    __builtin_prefetch(kp + 32 * HEAD, 0, 1);
    __builtin_prefetch(vb + 32, 0, 1);
    v8f s0 = (v8f){0.f,0.f,0.f,0.f,0.f,0.f,0.f,0.f};
    v8f s1 = s0;
    s0 = wmma_bf16(aq0, load_b_frag(kp, HEAD, lane), s0);
    s0 = wmma_bf16(aq1, load_b_frag(kp + 32, HEAD, lane), s0);
    s1 = wmma_bf16(aq0, load_b_frag(kp + 16 * HEAD, HEAD, lane), s1);
    s1 = wmma_bf16(aq1, load_b_frag(kp + 16 * HEAD + 32, HEAD, lane), s1);

    const bool tail = (kb + 32 > qbase);          // diagonal tile
#pragma unroll
    for (int i = 0; i < 8; ++i) {
      float sc0 = s0[i] * 0.125f;                 // 1/sqrt(64)
      float sc1 = s1[i] * 0.125f;
      if (tail) {
        const int row = qbase + i + 8 * hf;
        if (kb + n > row)      sc0 = -1.0e30f;
        if (kb + 16 + n > row) sc1 = -1.0e30f;
      }
      // row max across the 16 lanes of this half-wave
      float rm = fmaxf(sc0, sc1);
      rm = fmaxf(rm, __shfl_xor(rm, 1));
      rm = fmaxf(rm, __shfl_xor(rm, 2));
      rm = fmaxf(rm, __shfl_xor(rm, 4));
      rm = fmaxf(rm, __shfl_xor(rm, 8));
      const float mnew = fmaxf(mstat[i], rm);
      const float p0 = __expf(sc0 - mnew);
      const float p1 = __expf(sc1 - mnew);
      float rs = p0 + p1;
      rs += __shfl_xor(rs, 1);
      rs += __shfl_xor(rs, 2);
      rs += __shfl_xor(rs, 4);
      rs += __shfl_xor(rs, 8);
      const float scale = __expf(mstat[i] - mnew);
      lstat[i] = lstat[i] * scale + rs;
      mstat[i] = mnew;
#pragma unroll
      for (int g = 0; g < 4; ++g) o[g][i] *= scale;
      // transpose P (C layout) into LDS row-major 16x32
      pbuf[wave][(i + 8 * hf) * 32 + n]      = (bf16)p0;
      pbuf[wave][(i + 8 * hf) * 32 + 16 + n] = (bf16)p1;
    }
    asm volatile("s_wait_dscnt 0x0" ::: "memory");
    const v16bf ap = load_a_frag(pbuf[wave], 32, lane);
#pragma unroll
    for (int g = 0; g < 4; ++g)
      o[g] = wmma_bf16(ap, load_b_frag(vb + (size_t)g * 16 * SEQ, SEQ, lane), o[g]);
  }

  const int b = bh >> 4, h = bh & 15;
#pragma unroll
  for (int i = 0; i < 8; ++i) {
    const float inv = 1.0f / lstat[i];
    const size_t row = (size_t)(b * SEQ + qbase + i + 8 * hf);
    bf16* dst = Oo + row * D_MODEL + h * HEAD;
#pragma unroll
    for (int g = 0; g < 4; ++g) dst[g * 16 + n] = (bf16)(o[g][i] * inv);
  }
}

// ---------------------------------------------------------- O projection
// grid.x = 512 row tiles * 2 column halves = 1024, 256 threads.
// A-tile staging done by the Tensor Data Mover (TENSORcnt).
__global__ __launch_bounds__(256)
void oproj_kernel(const bf16* __restrict__ A, const bf16* __restrict__ Wo,
                  float* __restrict__ out) {
  __shared__ __align__(16) bf16 xs[16 * 32];
  const int r  = blockIdx.x & 511;
  const int hc = blockIdx.x >> 9;
  const int wave = threadIdx.x >> 5, lane = threadIdx.x & 31;
  const int colbase = hc * 512 + wave * 64;
  const uint32_t xs_lds = (uint32_t)(uintptr_t)xs;

  v8f acc[4];
#pragma unroll
  for (int g = 0; g < 4; ++g) acc[g] = (v8f){0.f,0.f,0.f,0.f,0.f,0.f,0.f,0.f};

  for (int k0 = 0; k0 < D_MODEL; k0 += 32) {
    __syncthreads();
    if (wave == 0) {
      // TDM: 16x32-element bf16 tile, row stride D_MODEL elements.
      tdm_load_tile_2d(xs_lds, A + (size_t)(r * 16) * D_MODEL + k0,
                       /*tensor_d0=*/D_MODEL, /*tensor_d1=*/ROWS,
                       /*tile_d0=*/32, /*tile_d1=*/16, /*stride0=*/D_MODEL);
      __builtin_amdgcn_s_wait_tensorcnt(0);
    }
    __syncthreads();

    v16bf a = load_a_frag(xs, 32, lane);
#pragma unroll
    for (int g = 0; g < 4; ++g) {
      const bf16* wb = Wo + (size_t)(colbase + g * 16) * D_MODEL + k0;
      acc[g] = wmma_bf16(a, load_b_frag(wb, D_MODEL, lane), acc[g]);
    }
    if (k0 + 32 < D_MODEL)
      __builtin_prefetch(Wo + (size_t)colbase * D_MODEL + k0 + 32, 0, 1);
  }

  const int n = lane & 15, hf = lane >> 4;
#pragma unroll
  for (int g = 0; g < 4; ++g)
#pragma unroll
    for (int i = 0; i < 8; ++i)
      out[(size_t)(r * 16 + i + 8 * hf) * D_MODEL + colbase + g * 16 + n] =
          acc[g][i];
}

// --------------------------------------------------------------- launcher
extern "C" void kernel_launch(void* const* d_in, const int* in_sizes, int n_in,
                              void* d_out, int out_size, void* d_ws,
                              size_t ws_size, hipStream_t stream) {
  const float* X  = (const float*)d_in[0];
  const float* Wq = (const float*)d_in[1];
  const float* Wk = (const float*)d_in[2];
  const float* Wv = (const float*)d_in[3];
  const float* Wo = (const float*)d_in[4];
  const long long* tp = (const long long*)d_in[5];
  float* out = (float*)d_out;

  char* ws = (char*)d_ws;
  size_t off = 0;
  const size_t XB = (size_t)ROWS * D_MODEL * sizeof(bf16);     // 16 MB
  const size_t WB = (size_t)D_MODEL * D_MODEL * sizeof(bf16);  //  2 MB
  bf16* Xb  = (bf16*)(ws + off); off += XB;
  bf16* Wqb = (bf16*)(ws + off); off += WB;
  bf16* Wkb = (bf16*)(ws + off); off += WB;
  bf16* Wvb = (bf16*)(ws + off); off += WB;
  bf16* Wob = (bf16*)(ws + off); off += WB;
  bf16* Qb  = (bf16*)(ws + off); off += XB;   // [B,H,S,64]
  bf16* Kb  = (bf16*)(ws + off); off += XB;   // [B,H,S,64]
  bf16* Vt  = (bf16*)(ws + off); off += XB;   // [B,H,64,S]
  bf16* Ab  = (bf16*)(ws + off); off += XB;   // attn out [8192,1024]

  const int nx4 = (ROWS * D_MODEL) / 4;
  const int nw4 = (D_MODEL * D_MODEL) / 4;
  f32_to_bf16_vec<<<1024, 256, 0, stream>>>(X,  Xb,  nx4);
  f32_to_bf16_vec<<<512,  256, 0, stream>>>(Wq, Wqb, nw4);
  f32_to_bf16_vec<<<512,  256, 0, stream>>>(Wk, Wkb, nw4);
  f32_to_bf16_vec<<<512,  256, 0, stream>>>(Wv, Wvb, nw4);
  f32_to_bf16_vec<<<512,  256, 0, stream>>>(Wo, Wob, nw4);

  qkv_rope_kernel<<<3072, 256, 0, stream>>>(Xb, Wqb, Wkb, Wvb, tp, Qb, Kb, Vt);
  flash_attn_kernel<<<2048, 128, 0, stream>>>(Qb, Kb, Vt, Ab);
  oproj_kernel<<<1024, 256, 0, stream>>>(Ab, Wob, out);
}
